// GNNEncoder_31842887532754
// MI455X (gfx1250) — matrix operand order repaired and checked by reading, compiled, not run on hardware
//
#include <hip/hip_runtime.h>

// ---------------------------------------------------------------------------
// GNN EdgeConv stack for MI455X (gfx1250): bf16 WMMA edge-MLPs + atomic-max
// segment aggregation.  Wave32; v_wmma_f32_16x16x32_bf16 is the core op.
// v2: 4 M-tiles (64 edges) per workgroup, B-fragment register reuse across
//     M-tiles (4 WMMA per weight fetch), vectorized A-tile gather.
// ---------------------------------------------------------------------------

typedef __attribute__((ext_vector_type(16))) __bf16 v16bf;
typedef __attribute__((ext_vector_type(8)))  __bf16 v8bf;
typedef __attribute__((ext_vector_type(4)))  __bf16 v4bf;
typedef __attribute__((ext_vector_type(8)))  float  v8f;
typedef __attribute__((ext_vector_type(4)))  float  v4f;

#define NNODES 65536
#define NEDGES 1048576
#define NGRAPH 64

// float -> bf16 round-to-nearest-even (bit manipulation; no __bf16 cvt dependency)
__device__ __forceinline__ __bf16 f2bf(float f) {
    unsigned u = __float_as_uint(f);
    unsigned r = (u + 0x7FFFu + ((u >> 16) & 1u)) >> 16;
    unsigned short s = (unsigned short)r;
    return __builtin_bit_cast(__bf16, s);
}

// Monotonic-int trick for float atomic max (works with -inf init, mixed signs)
__device__ __forceinline__ void atomicMaxF(float* addr, float val) {
    if (val >= 0.0f) atomicMax((int*)addr, __float_as_int(val));
    else             atomicMin((unsigned int*)addr, __float_as_uint(val));
}

__device__ __forceinline__ bool notFinite(float v) {
    return (__float_as_uint(v) & 0x7F800000u) == 0x7F800000u;
}

// ---------------------------------------------------------------------------
__global__ void fill_neginf_kernel(float* __restrict__ p, int n) {
    int i = blockIdx.x * blockDim.x + threadIdx.x;
    if (i < n) p[i] = -__builtin_inff();
}

__global__ void finalize_max_kernel(float* __restrict__ p, int n) {
    int i = blockIdx.x * blockDim.x + threadIdx.x;
    if (i < n) {
        float v = p[i];
        if (notFinite(v)) p[i] = 0.0f;
    }
}

// Convert weight [Kin x F] (row major, as in reference) to bf16 TRANSPOSED
// layout wT[n*Kpad + k], zero-padding K up to Kpad.  Transposed layout makes
// each lane's WMMA B-fragment a contiguous 32-byte run.
__global__ void cvt_weightT_kernel(const float* __restrict__ w,
                                   __bf16* __restrict__ wT,
                                   int Kin, int Kpad, int F) {
    int i = blockIdx.x * blockDim.x + threadIdx.x;
    if (i >= Kpad * F) return;
    int n = i / Kpad;
    int k = i % Kpad;
    float v = (k < Kin) ? w[k * F + n] : 0.0f;
    wT[i] = f2bf(v);
}

// ---------------------------------------------------------------------------
// Fused EdgeConv layer, MTILES*16 edges per workgroup:
//   A = cat[x_dst, x_src - x_dst]   (MTILES*16 x K1PAD, bf16, LDS)
//   H = relu(A @ W1 + b1)           (MTILES*16 x F, bf16, LDS)   -- WMMA bf16
//   O = H @ W2 + b2                 (f32 accumulators)           -- WMMA bf16
//   xout[dst, :] = atomic max with O rows
// One wave owns a 16-column output tile (strided over F/16); for each k-step
// the B fragment is loaded once and reused for all MTILES WMMAs.
// ---------------------------------------------------------------------------
template<int FIN, int K1PAD, int F, int NWAVES, int MTILES>
__global__ __launch_bounds__(NWAVES * 32)
void edgeconv_wmma_kernel(const float* __restrict__ xin,
                          const int*   __restrict__ src,
                          const int*   __restrict__ dst,
                          const __bf16* __restrict__ w1T,
                          const float* __restrict__ b1,
                          const __bf16* __restrict__ w2T,
                          const float* __restrict__ b2,
                          float* __restrict__ xout) {
    constexpr int M = MTILES * 16;                 // edges per workgroup
    __shared__ __align__(16) __bf16 Atile[M * K1PAD];
    __shared__ __align__(16) __bf16 Htile[M * F];
    __shared__ int dsts[M];
    __shared__ int srcs[M];

    const int tid = threadIdx.x;
    const int e0  = blockIdx.x * M;

    if (tid < M) {
        dsts[tid] = dst[e0 + tid];
        srcs[tid] = src[e0 + tid];
    }
    __syncthreads();

    // Cooperative build of A tile: cat[xi, xj - xi], zero-padded to K1PAD.
    if constexpr ((FIN % 4) == 0) {
        // vector path: 4 consecutive K per thread -> float4 gathers, b64 LDS store
        for (int idx = tid; idx < M * (K1PAD / 4); idx += NWAVES * 32) {
            int m  = idx / (K1PAD / 4);
            int k4 = (idx % (K1PAD / 4)) * 4;
            v4f val = {};
            if (k4 < FIN) {
                val = *(const v4f*)&xin[(size_t)dsts[m] * FIN + k4];
            } else if (k4 < 2 * FIN) {
                int kk = k4 - FIN;
                v4f xj = *(const v4f*)&xin[(size_t)srcs[m] * FIN + kk];
                v4f xi = *(const v4f*)&xin[(size_t)dsts[m] * FIN + kk];
                val = xj - xi;
            }
            v4bf bv;
            #pragma unroll
            for (int t = 0; t < 4; ++t) bv[t] = f2bf(val[t]);
            *(v4bf*)&Atile[m * K1PAD + k4] = bv;
        }
    } else {
        for (int idx = tid; idx < M * K1PAD; idx += NWAVES * 32) {
            int m = idx / K1PAD;
            int k = idx % K1PAD;
            float v = 0.0f;
            if (k < FIN) {
                v = xin[(size_t)dsts[m] * FIN + k];
            } else if (k < 2 * FIN) {
                int kk = k - FIN;
                v = xin[(size_t)srcs[m] * FIN + kk] - xin[(size_t)dsts[m] * FIN + kk];
            }
            Atile[m * K1PAD + k] = f2bf(v);
        }
    }
    __syncthreads();

    const int lane = tid & 31;
    const int wave = tid >> 5;
    const int mrow = lane & 15;   // A row (within tile) / B column / D column
    const int hi   = lane >> 4;   // which half of the wave

    // ---- GEMM1: H = relu(A @ W1 + b1) ----
    for (int nt = wave; nt < F / 16; nt += NWAVES) {
        v8f acc[MTILES];
        #pragma unroll
        for (int mt = 0; mt < MTILES; ++mt) acc[mt] = v8f{};
        const int ncol = nt * 16 + mrow;
        #pragma unroll
        for (int k0 = 0; k0 < K1PAD; k0 += 32) {
            // B fragment once per k-step (transposed weights: 16 contiguous K)
            const __bf16* bp = &w1T[(size_t)ncol * K1PAD + k0 + hi * 16];
            v8bf blo = *(const v8bf*)(bp);
            v8bf bhi = *(const v8bf*)(bp + 8);
            v16bf b = __builtin_shufflevector(blo, bhi,
                0, 1, 2, 3, 4, 5, 6, 7, 8, 9, 10, 11, 12, 13, 14, 15);
            #pragma unroll
            for (int mt = 0; mt < MTILES; ++mt) {
                const __bf16* ap = &Atile[(mt * 16 + mrow) * K1PAD + k0 + hi * 16];
                v8bf alo = *(const v8bf*)(ap);
                v8bf ahi = *(const v8bf*)(ap + 8);
                v16bf a = __builtin_shufflevector(alo, ahi,
                    0, 1, 2, 3, 4, 5, 6, 7, 8, 9, 10, 11, 12, 13, 14, 15);
                acc[mt] = __builtin_amdgcn_wmma_f32_16x16x32_bf16(
                    false, a, false, b, (short)0, acc[mt], false, false);
            }
        }
        const float bias = b1[ncol];
        #pragma unroll
        for (int mt = 0; mt < MTILES; ++mt) {
            #pragma unroll
            for (int v = 0; v < 8; ++v) {
                int m = mt * 16 + v + hi * 8;
                float hval = acc[mt][v] + bias;
                hval = hval > 0.0f ? hval : 0.0f;       // relu
                Htile[m * F + ncol] = f2bf(hval);
            }
        }
    }
    __syncthreads();

    // ---- GEMM2: O = H @ W2 + b2, scatter atomic-max into xout ----
    for (int nt = wave; nt < F / 16; nt += NWAVES) {
        v8f acc[MTILES];
        #pragma unroll
        for (int mt = 0; mt < MTILES; ++mt) acc[mt] = v8f{};
        const int ncol = nt * 16 + mrow;
        #pragma unroll
        for (int k0 = 0; k0 < F; k0 += 32) {
            const __bf16* bp = &w2T[(size_t)ncol * F + k0 + hi * 16];
            v8bf blo = *(const v8bf*)(bp);
            v8bf bhi = *(const v8bf*)(bp + 8);
            v16bf b = __builtin_shufflevector(blo, bhi,
                0, 1, 2, 3, 4, 5, 6, 7, 8, 9, 10, 11, 12, 13, 14, 15);
            #pragma unroll
            for (int mt = 0; mt < MTILES; ++mt) {
                const __bf16* ap = &Htile[(mt * 16 + mrow) * F + k0 + hi * 16];
                v8bf alo = *(const v8bf*)(ap);
                v8bf ahi = *(const v8bf*)(ap + 8);
                v16bf a = __builtin_shufflevector(alo, ahi,
                    0, 1, 2, 3, 4, 5, 6, 7, 8, 9, 10, 11, 12, 13, 14, 15);
                acc[mt] = __builtin_amdgcn_wmma_f32_16x16x32_bf16(
                    false, a, false, b, (short)0, acc[mt], false, false);
            }
        }
        const float bias = b2[ncol];
        #pragma unroll
        for (int mt = 0; mt < MTILES; ++mt) {
            #pragma unroll
            for (int v = 0; v < 8; ++v) {
                int m = mt * 16 + v + hi * 8;
                float o = acc[mt][v] + bias;
                atomicMaxF(&xout[(size_t)dsts[m] * F + ncol], o);
            }
        }
    }
}

// ---------------------------------------------------------------------------
__global__ void pool_max_kernel(const float* __restrict__ x3,
                                const int* __restrict__ batch,
                                float* __restrict__ pooled) {
    int i = blockIdx.x * blockDim.x + threadIdx.x;   // over N*256
    if (i >= NNODES * 256) return;
    int n = i >> 8;
    int c = i & 255;
    int g = batch[n];
    atomicMaxF(&pooled[g * 256 + c], x3[i]);
}

__global__ void fc_kernel(const float* __restrict__ pooled,
                          const float* __restrict__ w_fc,
                          const float* __restrict__ b_fc,
                          float* __restrict__ out) {
    int g = blockIdx.x;     // 64 graphs
    int o = threadIdx.x;    // 128 outputs
    float s = b_fc[o];
    #pragma unroll 4
    for (int c = 0; c < 256; ++c) {
        float p = pooled[g * 256 + c];
        if (notFinite(p)) p = 0.0f;       // empty graph -> 0
        s += p * w_fc[c * 128 + o];
    }
    out[g * 128 + o] = s;
}

// ---------------------------------------------------------------------------
extern "C" void kernel_launch(void* const* d_in, const int* in_sizes, int n_in,
                              void* d_out, int out_size, void* d_ws, size_t ws_size,
                              hipStream_t stream) {
    (void)in_sizes; (void)n_in; (void)out_size; (void)ws_size;

    const float* x     = (const float*)d_in[0];
    const int*   ei    = (const int*)  d_in[1];
    const int*   batch = (const int*)  d_in[2];
    const float* w1_1  = (const float*)d_in[3];
    const float* b1_1  = (const float*)d_in[4];
    const float* w2_1  = (const float*)d_in[5];
    const float* b2_1  = (const float*)d_in[6];
    const float* w1_2  = (const float*)d_in[7];
    const float* b1_2  = (const float*)d_in[8];
    const float* w2_2  = (const float*)d_in[9];
    const float* b2_2  = (const float*)d_in[10];
    const float* w1_3  = (const float*)d_in[11];
    const float* b1_3  = (const float*)d_in[12];
    const float* w2_3  = (const float*)d_in[13];
    const float* b2_3  = (const float*)d_in[14];
    const float* w_fc  = (const float*)d_in[15];
    const float* b_fc  = (const float*)d_in[16];

    const int* src = ei;             // edge_index[0] = x_j source
    const int* dst = ei + NEDGES;    // edge_index[1] = x_i target / aggregation

    // ---- workspace carve-up (256B aligned) ----
    char*  ws  = (char*)d_ws;
    size_t off = 0;
    auto carve = [&](size_t bytes) -> void* {
        void* p = ws + off;
        off = (off + bytes + 255) & ~(size_t)255;
        return p;
    };
    float*  x1     = (float*)carve((size_t)NNODES * 64  * sizeof(float));
    float*  x2     = (float*)carve((size_t)NNODES * 128 * sizeof(float));
    float*  x3     = (float*)carve((size_t)NNODES * 256 * sizeof(float));
    float*  pooled = (float*)carve((size_t)NGRAPH * 256 * sizeof(float));
    __bf16* w11T   = (__bf16*)carve((size_t)64  * 32  * sizeof(__bf16)); // pad K 6->32
    __bf16* w21T   = (__bf16*)carve((size_t)64  * 64  * sizeof(__bf16));
    __bf16* w12T   = (__bf16*)carve((size_t)128 * 128 * sizeof(__bf16));
    __bf16* w22T   = (__bf16*)carve((size_t)128 * 128 * sizeof(__bf16));
    __bf16* w13T   = (__bf16*)carve((size_t)256 * 256 * sizeof(__bf16));
    __bf16* w23T   = (__bf16*)carve((size_t)256 * 256 * sizeof(__bf16));

    // ---- weight conversion (tiny, once per launch) ----
    auto cvtGrid = [](int n) { return (n + 255) / 256; };
    cvt_weightT_kernel<<<cvtGrid(64 * 32),   256, 0, stream>>>(w1_1, w11T,   6,  32,  64);
    cvt_weightT_kernel<<<cvtGrid(64 * 64),   256, 0, stream>>>(w2_1, w21T,  64,  64,  64);
    cvt_weightT_kernel<<<cvtGrid(128 * 128), 256, 0, stream>>>(w1_2, w12T, 128, 128, 128);
    cvt_weightT_kernel<<<cvtGrid(128 * 128), 256, 0, stream>>>(w2_2, w22T, 128, 128, 128);
    cvt_weightT_kernel<<<cvtGrid(256 * 256), 256, 0, stream>>>(w1_3, w13T, 256, 256, 256);
    cvt_weightT_kernel<<<cvtGrid(256 * 256), 256, 0, stream>>>(w2_3, w23T, 256, 256, 256);

    constexpr int MT = 4;                         // 64 edges per workgroup
    const int edgeBlocks = NEDGES / (16 * MT);

    // ---- layer 1: 3 -> 64 ----
    fill_neginf_kernel<<<cvtGrid(NNODES * 64), 256, 0, stream>>>(x1, NNODES * 64);
    edgeconv_wmma_kernel<3, 32, 64, 4, MT><<<edgeBlocks, 4 * 32, 0, stream>>>(
        x, src, dst, w11T, b1_1, w21T, b2_1, x1);
    finalize_max_kernel<<<cvtGrid(NNODES * 64), 256, 0, stream>>>(x1, NNODES * 64);

    // ---- layer 2: 64 -> 128 ----
    fill_neginf_kernel<<<cvtGrid(NNODES * 128), 256, 0, stream>>>(x2, NNODES * 128);
    edgeconv_wmma_kernel<64, 128, 128, 8, MT><<<edgeBlocks, 8 * 32, 0, stream>>>(
        x1, src, dst, w12T, b1_2, w22T, b2_2, x2);
    finalize_max_kernel<<<cvtGrid(NNODES * 128), 256, 0, stream>>>(x2, NNODES * 128);

    // ---- layer 3: 128 -> 256 ----
    fill_neginf_kernel<<<cvtGrid(NNODES * 256), 256, 0, stream>>>(x3, NNODES * 256);
    edgeconv_wmma_kernel<128, 256, 256, 8, MT><<<edgeBlocks, 8 * 32, 0, stream>>>(
        x2, src, dst, w13T, b1_3, w23T, b2_3, x3);
    finalize_max_kernel<<<cvtGrid(NNODES * 256), 256, 0, stream>>>(x3, NNODES * 256);

    // ---- graph max-pool + FC ----
    fill_neginf_kernel<<<cvtGrid(NGRAPH * 256), 256, 0, stream>>>(pooled, NGRAPH * 256);
    pool_max_kernel<<<cvtGrid(NNODES * 256), 256, 0, stream>>>(x3, batch, pooled);
    fc_kernel<<<NGRAPH, 128, 0, stream>>>(pooled, w_fc, b_fc, (float*)d_out);
}
